// NanoGPTLanguageModel_36807869727473
// MI455X (gfx1250) — compile-verified, matrix operand order checked by reading, not run on hardware
//
#include <hip/hip_runtime.h>
#include <hip/hip_bf16.h>

// ---- model dims ----
#define NLAYER 6
#define NHEAD  6
#define CEMB   384
#define TSEQ   256
#define VOCAB  65
#define DHEAD  64
#define FFDIM  1536
#define BATCH  256
#define BT     (BATCH * TSEQ)

typedef __attribute__((ext_vector_type(16))) __bf16 v16bf;
typedef __attribute__((ext_vector_type(8)))  float  v8f;

// TDM descriptor SGPR groups (6-arg builtin form: g0, g1, g2, g3, g4, cpol)
typedef __attribute__((ext_vector_type(4))) unsigned int tdm_g0_t;
typedef __attribute__((ext_vector_type(8))) int          tdm_g1_t;
typedef __attribute__((ext_vector_type(4))) int          tdm_g2_t;

union AF { v16bf v; unsigned int u[8]; };

// fp32 -> bf16 round-to-nearest-even (raw ushort bits)
__device__ __forceinline__ unsigned short f2b(float f) {
    unsigned int u = __float_as_uint(f);
    unsigned int r = (u + 0x7FFFu + ((u >> 16) & 1u)) >> 16;
    return (unsigned short)r;
}

// ---------------- elementwise fp32 -> bf16 convert (weights) ----------------
__global__ void cvt_kernel(const float* __restrict__ s, unsigned short* __restrict__ d, size_t n) {
    size_t i = (size_t)blockIdx.x * blockDim.x + threadIdx.x;
    size_t stride = (size_t)gridDim.x * blockDim.x;
    for (; i < n; i += stride) d[i] = f2b(s[i]);
}

// ---------------- embedding: x = tok_emb[idx] + pos_emb ----------------
__global__ void embed_kernel(const int* __restrict__ idx, const float* __restrict__ tok,
                             const float* __restrict__ pos, float* __restrict__ x) {
    int row = blockIdx.x;
    int t = row % TSEQ;
    int tk = idx[row];
    for (int c = threadIdx.x; c < CEMB; c += blockDim.x)
        x[(size_t)row * CEMB + c] = tok[(size_t)tk * CEMB + c] + pos[(size_t)t * CEMB + c];
}

// ---------------- layernorm (fp32 in -> bf16 out) ----------------
__global__ __launch_bounds__(128) void ln_kernel(const float* __restrict__ x,
                                                 const float* __restrict__ g,
                                                 const float* __restrict__ bta,
                                                 unsigned short* __restrict__ dst) {
    __shared__ float red[128];
    const int row = blockIdx.x, tid = threadIdx.x;
    const float* xr = x + (size_t)row * CEMB;
    float s = 0.f;
    for (int c = tid; c < CEMB; c += 128) s += xr[c];
    red[tid] = s; __syncthreads();
    for (int o = 64; o > 0; o >>= 1) { if (tid < o) red[tid] += red[tid + o]; __syncthreads(); }
    float mean = red[0] * (1.0f / CEMB);
    __syncthreads();
    float v = 0.f;
    for (int c = tid; c < CEMB; c += 128) { float d2 = xr[c] - mean; v += d2 * d2; }
    red[tid] = v; __syncthreads();
    for (int o = 64; o > 0; o >>= 1) { if (tid < o) red[tid] += red[tid + o]; __syncthreads(); }
    float rstd = rsqrtf(red[0] * (1.0f / CEMB) + 1e-5f);
    unsigned short* dr = dst + (size_t)row * CEMB;
    for (int c = tid; c < CEMB; c += 128)
        dr[c] = f2b((xr[c] - mean) * rstd * g[c] + bta[c]);
}

// ---------------- tiled bf16 WMMA GEMM with TDM double-buffered A tiles ----------------
// D[M,N] = A[M,K](bf16) * Bw[K,N](bf16) (+bias)(+relu)
// outf: fp32 output (addRes ? += : =), may be null; outb: bf16 output, may be null
// block tile: 256(M) x 64(N), K-step 32, 8 waves; each wave: 32 rows x 64 cols = 8 WMMA tiles
__global__ __launch_bounds__(256) void gemm_bf16_kernel(
    const unsigned short* __restrict__ A, const unsigned short* __restrict__ Bw,
    const float* __restrict__ bias, float* __restrict__ outf,
    unsigned short* __restrict__ outb,
    int M, int N, int K, int addRes, int doRelu) {
    __shared__ __align__(16) unsigned short As[2][256 * 32];   // [buf][row][k]  (TDM-filled)
    __shared__ __align__(16) unsigned short Bts[2][64 * 32];   // [buf][n][k]   (transposed)
    const int m0 = blockIdx.y * 256;
    const int n0 = blockIdx.x * 64;
    const int tid = threadIdx.x;
    const int wave = tid >> 5, lane = tid & 31;
    const int kadd = (lane >> 4) << 3;      // ISA 16-bit operand lane striping
    const int nK = K >> 5;

    // stage B tile (32 k x 64 n) transposed into Bts[buf][n][k]
    auto stageB = [&](int kk, int buf) {
        if ((N & 63) == 0) {
            int k = tid >> 3, ng = tid & 7;
            uint4 d = *(const uint4*)(Bw + (size_t)(kk + k) * N + n0 + ng * 8);
            unsigned int dd[4] = {d.x, d.y, d.z, d.w};
            #pragma unroll
            for (int j = 0; j < 4; ++j) {
                Bts[buf][(ng * 8 + j * 2    ) * 32 + k] = (unsigned short)(dd[j] & 0xffffu);
                Bts[buf][(ng * 8 + j * 2 + 1) * 32 + k] = (unsigned short)(dd[j] >> 16);
            }
        } else {  // ragged N (vocab head): scalar, zero-pad
            #pragma unroll
            for (int i = 0; i < 8; ++i) {
                int e = tid * 8 + i;
                int k = e >> 6, nl = e & 63, n = n0 + nl;
                Bts[buf][nl * 32 + k] = (n < N) ? Bw[(size_t)(kk + k) * N + n] : (unsigned short)0;
            }
        }
    };

    // async DMA of A tile (256 rows x 32 k) into As[buf] via Tensor Data Mover
    auto tdmA = [&](int kk, int buf) {
        unsigned long long ga = (unsigned long long)(size_t)(A + (size_t)m0 * K + kk);
        unsigned lds = (unsigned)(size_t)&As[buf][0];
        tdm_g0_t g0;
        g0.x = 1u;                                                    // count=1, user desc
        g0.y = lds;                                                   // lds_addr
        g0.z = (unsigned)(ga & 0xffffffffull);                        // global_addr[31:0]
        g0.w = (unsigned)((ga >> 32) & 0x01ffffffull) | (2u << 30);   // global_addr[56:32], type=2
        tdm_g1_t g1;
        g1[0] = (int)(1u << 16);                                      // data_size=1 (2 bytes)
        g1[1] = (int)(((unsigned)K & 0xffffu) << 16);                 // tensor_dim0[15:0]
        g1[2] = (int)((((unsigned)K >> 16) & 0xffffu) |
                      (((unsigned)M & 0xffffu) << 16));               // dim0[31:16], dim1[15:0]
        g1[3] = (int)((((unsigned)M >> 16) & 0xffffu) | (32u << 16)); // dim1[31:16], tile_dim0=32
        g1[4] = (int)256u;                                            // tile_dim1=256, tile_dim2=0
        g1[5] = (int)(unsigned)K;                                     // tensor_dim0_stride[31:0]
        g1[6] = 0;                                                    // stride0[47:32], stride1 lo
        g1[7] = 0;
        tdm_g2_t gz4 = {0, 0, 0, 0};                                  // 2-D: groups 2/3 unused
        tdm_g1_t gz8 = {0, 0, 0, 0, 0, 0, 0, 0};
        __builtin_amdgcn_tensor_load_to_lds(g0, g1, gz4, gz4, gz8, 0);
    };

    v8f acc[2][4];
    #pragma unroll
    for (int rb = 0; rb < 2; ++rb)
        #pragma unroll
        for (int c = 0; c < 4; ++c)
            acc[rb][c] = (v8f){0.f,0.f,0.f,0.f,0.f,0.f,0.f,0.f};

    // prologue: tile 0 into buffer 0
    stageB(0, 0);
    if (wave == 0) tdmA(0, 0);

    int buf = 0;
    for (int kt = 0; kt < nK; ++kt) {
        const int kk = kt << 5;
        // prefetch next tile into other buffer (DMA overlaps compute of current tile)
        if (kt + 1 < nK) {
            stageB(kk + 32, buf ^ 1);
            if (wave == 0) tdmA(kk + 32, buf ^ 1);
        }
        // in-order TENSORcnt: <=1 outstanding means current tile's DMA is complete
        if (wave == 0) {
            if (kt + 1 < nK) __builtin_amdgcn_s_wait_tensorcnt((short)1);
            else             __builtin_amdgcn_s_wait_tensorcnt((short)0);
        }
        __syncthreads();   // tile kt (TDM A + staged B) visible to all waves

        AF af[2];
        #pragma unroll
        for (int rb = 0; rb < 2; ++rb) {
            int rowl = wave * 32 + rb * 16 + (lane & 15);
            #pragma unroll
            for (int j = 0; j < 8; ++j) {
                int k = ((j >> 2) << 4) + kadd + ((j & 3) << 1);
                af[rb].u[j] = *(const unsigned int*)&As[buf][rowl * 32 + k];
            }
        }
        #pragma unroll
        for (int c = 0; c < 4; ++c) {
            AF bf;
            int nl = c * 16 + (lane & 15);
            #pragma unroll
            for (int j = 0; j < 8; ++j) {
                int k = ((j >> 2) << 4) + kadd + ((j & 3) << 1);
                bf.u[j] = *(const unsigned int*)&Bts[buf][nl * 32 + k];
            }
            acc[0][c] = __builtin_amdgcn_wmma_f32_16x16x32_bf16(
                false, af[0].v, false, bf.v, (short)0, acc[0][c], false, false);
            acc[1][c] = __builtin_amdgcn_wmma_f32_16x16x32_bf16(
                false, af[1].v, false, bf.v, (short)0, acc[1][c], false, false);
        }
        __syncthreads();   // compute(kt) done before next iteration overwrites buf
        buf ^= 1;
    }

    // epilogue (C/D layout: VGPR r, lane L -> M = r + 8*(L/16), N = L%16)
    #pragma unroll
    for (int rb = 0; rb < 2; ++rb) {
        #pragma unroll
        for (int c = 0; c < 4; ++c) {
            #pragma unroll
            for (int r = 0; r < 8; ++r) {
                int row = m0 + wave * 32 + rb * 16 + r + ((lane >> 4) << 3);
                int col = n0 + c * 16 + (lane & 15);
                if (col < N) {
                    float v = acc[rb][c][r];
                    if (bias) v += bias[col];
                    if (doRelu) v = fmaxf(v, 0.0f);
                    size_t o = (size_t)row * N + col;
                    if (outf) { if (addRes) outf[o] += v; else outf[o] = v; }
                    if (outb) outb[o] = f2b(v);
                }
            }
        }
    }
}

// ---------------- attention: one block per (b, h, 32-query block) ----------------
// q/k/v/o are bf16 [BT, C] with head slice at column h*DHEAD
__global__ __launch_bounds__(256) void attn_kernel(
    const unsigned short* __restrict__ qp, const unsigned short* __restrict__ kp,
    const unsigned short* __restrict__ vp, unsigned short* __restrict__ op) {
    __shared__ __align__(16) unsigned short Qs[32 * 64];    // [qrow][dh]
    __shared__ __align__(16) unsigned short KV[64 * 64];    // K tile [key][dh] / Vt tile [dh][key]
    __shared__ float S[32 * 256];                           // score rows
    __shared__ unsigned short Abf[32 * 256];                // softmax probs, bf16

    const int qblk = blockIdx.x, h = blockIdx.y, b = blockIdx.z;
    const int q0 = qblk * 32;
    const int tid = threadIdx.x, wave = tid >> 5, lane = tid & 31;
    const int kadd = (lane >> 4) << 3;
    const size_t hoff = (size_t)h * DHEAD;
    const int rb = wave >> 2;       // query row-block (0..1)
    const int ct = wave & 3;        // column tile (0..3)

    // load Q block 32x64
    {
        int row = tid >> 3, seg = tid & 7;
        *(uint4*)&Qs[row * 64 + seg * 8] =
            *(const uint4*)(qp + (size_t)(b * TSEQ + q0 + row) * CEMB + hoff + seg * 8);
    }

    // ---- scores S = QK^T * Dh^-0.5, causal mask ----
    for (int kt = 0; kt < 4; ++kt) {
        #pragma unroll
        for (int i = 0; i < 2; ++i) {
            int id = tid * 2 + i;
            int row = id >> 3, seg = id & 7;
            *(uint4*)&KV[row * 64 + seg * 8] =
                *(const uint4*)(kp + (size_t)(b * TSEQ + kt * 64 + row) * CEMB + hoff + seg * 8);
        }
        __syncthreads();
        v8f acc = (v8f){0.f,0.f,0.f,0.f,0.f,0.f,0.f,0.f};
        #pragma unroll
        for (int ks = 0; ks < 2; ++ks) {
            AF af, bf;
            int rowl = rb * 16 + (lane & 15);
            int nl = ct * 16 + (lane & 15);
            #pragma unroll
            for (int j = 0; j < 8; ++j) {
                int k = ks * 32 + ((j >> 2) << 4) + kadd + ((j & 3) << 1);
                af.u[j] = *(const unsigned int*)&Qs[rowl * 64 + k];
                bf.u[j] = *(const unsigned int*)&KV[nl * 64 + k];
            }
            acc = __builtin_amdgcn_wmma_f32_16x16x32_bf16(
                false, af.v, false, bf.v, (short)0, acc, false, false);
        }
        #pragma unroll
        for (int r = 0; r < 8; ++r) {
            int row = rb * 16 + r + ((lane >> 4) << 3);
            int colg = kt * 64 + ct * 16 + (lane & 15);
            float v = acc[r] * 0.125f;                 // Dh^-0.5 = 1/8
            if (colg > q0 + row) v = -1e30f;           // causal
            S[row * 256 + colg] = v;
        }
        __syncthreads();
    }

    // ---- row softmax -> bf16 (8 lanes per row, same wave, shfl_xor reduce) ----
    {
        int row = tid >> 3, sub = tid & 7;
        float* sr = &S[row * 256];
        float m = -1e30f;
        for (int c = sub; c < 256; c += 8) m = fmaxf(m, sr[c]);
        #pragma unroll
        for (int o = 4; o >= 1; o >>= 1) m = fmaxf(m, __shfl_xor(m, o, 32));
        float s = 0.f;
        for (int c = sub; c < 256; c += 8) { float p = __expf(sr[c] - m); sr[c] = p; s += p; }
        #pragma unroll
        for (int o = 4; o >= 1; o >>= 1) s += __shfl_xor(s, o, 32);
        float inv = 1.0f / s;
        for (int c = sub; c < 256; c += 8) Abf[row * 256 + c] = f2b(sr[c] * inv);
    }
    __syncthreads();

    // ---- O = att @ V ----
    const int dt = wave & 3;        // dh tile (0..3)
    v8f oacc = (v8f){0.f,0.f,0.f,0.f,0.f,0.f,0.f,0.f};
    for (int kt = 0; kt < 4; ++kt) {
        #pragma unroll
        for (int i = 0; i < 2; ++i) {                    // stage Vt[dh][key]
            int id = tid * 2 + i;
            int key = id >> 3, seg = id & 7;
            uint4 d = *(const uint4*)(vp + (size_t)(b * TSEQ + kt * 64 + key) * CEMB + hoff + seg * 8);
            unsigned int dd[4] = {d.x, d.y, d.z, d.w};
            #pragma unroll
            for (int j = 0; j < 4; ++j) {
                KV[(seg * 8 + j * 2    ) * 64 + key] = (unsigned short)(dd[j] & 0xffffu);
                KV[(seg * 8 + j * 2 + 1) * 64 + key] = (unsigned short)(dd[j] >> 16);
            }
        }
        __syncthreads();
        #pragma unroll
        for (int ks = 0; ks < 2; ++ks) {
            AF af, bf;
            int rowl = rb * 16 + (lane & 15);
            int nl = dt * 16 + (lane & 15);
            #pragma unroll
            for (int j = 0; j < 8; ++j) {
                int k = ((j >> 2) << 4) + kadd + ((j & 3) << 1);
                af.u[j] = *(const unsigned int*)&Abf[rowl * 256 + kt * 64 + ks * 32 + k];
                bf.u[j] = *(const unsigned int*)&KV[nl * 64 + ks * 32 + k];
            }
            oacc = __builtin_amdgcn_wmma_f32_16x16x32_bf16(
                false, af.v, false, bf.v, (short)0, oacc, false, false);
        }
        __syncthreads();
    }
    #pragma unroll
    for (int r = 0; r < 8; ++r) {
        int row = rb * 16 + r + ((lane >> 4) << 3);
        int dh = dt * 16 + (lane & 15);
        op[(size_t)(b * TSEQ + q0 + row) * CEMB + hoff + dh] = f2b(oacc[r]);
    }
}

// ---------------- loss ----------------
__global__ void loss_init_kernel(float* acc) { if (threadIdx.x == 0) *acc = 0.0f; }

__global__ void loss_row_kernel(const float* __restrict__ logits, const int* __restrict__ tgt,
                                float* __restrict__ acc, int rows) {
    int row = blockIdx.x * blockDim.x + threadIdx.x;
    if (row >= rows) return;
    const float* lr = logits + (size_t)row * VOCAB;
    float m = -1e30f;
    for (int c = 0; c < VOCAB; ++c) m = fmaxf(m, lr[c]);
    float s = 0.f;
    for (int c = 0; c < VOCAB; ++c) s += __expf(lr[c] - m);
    float lp = lr[tgt[row]] - m - __logf(s);
    atomicAdd(acc, lp);
}

__global__ void loss_final_kernel(const float* __restrict__ acc, float* __restrict__ out) {
    if (threadIdx.x == 0) *out = -(*acc) * (1.0f / BT);
}

// ---------------- host orchestration ----------------
extern "C" void kernel_launch(void* const* d_in, const int* in_sizes, int n_in,
                              void* d_out, int out_size, void* d_ws, size_t ws_size,
                              hipStream_t stream) {
    (void)in_sizes; (void)n_in; (void)out_size; (void)ws_size;
    const int*   idx  = (const int*)d_in[0];
    const int*   tgt  = (const int*)d_in[1];
    const float* tok  = (const float*)d_in[2];
    const float* pos  = (const float*)d_in[3];
    const float* Wq   = (const float*)d_in[4];
    const float* Wk   = (const float*)d_in[5];
    const float* Wv   = (const float*)d_in[6];
    const float* Wo   = (const float*)d_in[7];
    const float* bo   = (const float*)d_in[8];
    const float* ln1g = (const float*)d_in[9];
    const float* ln1b = (const float*)d_in[10];
    const float* ln2g = (const float*)d_in[11];
    const float* ln2b = (const float*)d_in[12];
    const float* W1   = (const float*)d_in[13];
    const float* b1   = (const float*)d_in[14];
    const float* W2   = (const float*)d_in[15];
    const float* b2   = (const float*)d_in[16];
    const float* lnfg = (const float*)d_in[17];
    const float* lnfb = (const float*)d_in[18];
    const float* Wh   = (const float*)d_in[19];
    const float* bh   = (const float*)d_in[20];

    char* base = (char*)d_ws;
    size_t off = 0;
    auto alloc = [&](size_t bytes) -> void* {
        void* p = base + off;
        off += (bytes + 255) & ~(size_t)255;
        return p;
    };
    float*          xf  = (float*)alloc((size_t)BT * CEMB * 4);          // fp32 residual stream
    unsigned short* hb  = (unsigned short*)alloc((size_t)BT * CEMB * 2); // ln output (reused)
    unsigned short* qbf = (unsigned short*)alloc((size_t)BT * CEMB * 2);
    unsigned short* kbf = (unsigned short*)alloc((size_t)BT * CEMB * 2);
    unsigned short* vbf = (unsigned short*)alloc((size_t)BT * CEMB * 2);
    unsigned short* obf = (unsigned short*)alloc((size_t)BT * CEMB * 2);
    unsigned short* a1b = (unsigned short*)alloc((size_t)BT * FFDIM * 2);
    unsigned short* wqB = (unsigned short*)alloc((size_t)NLAYER * CEMB * CEMB * 2);
    unsigned short* wkB = (unsigned short*)alloc((size_t)NLAYER * CEMB * CEMB * 2);
    unsigned short* wvB = (unsigned short*)alloc((size_t)NLAYER * CEMB * CEMB * 2);
    unsigned short* woB = (unsigned short*)alloc((size_t)NLAYER * CEMB * CEMB * 2);
    unsigned short* w1B = (unsigned short*)alloc((size_t)NLAYER * CEMB * FFDIM * 2);
    unsigned short* w2B = (unsigned short*)alloc((size_t)NLAYER * FFDIM * CEMB * 2);
    unsigned short* whB = (unsigned short*)alloc((size_t)CEMB * VOCAB * 2);
    float* lossAcc = (float*)alloc(sizeof(float));

    auto cvt = [&](const float* s, unsigned short* d, size_t n) {
        int blocks = (int)((n + 2047) / 2048);
        if (blocks > 8192) blocks = 8192;
        cvt_kernel<<<blocks, 256, 0, stream>>>(s, d, n);
    };
    cvt(Wq, wqB, (size_t)NLAYER * CEMB * CEMB);
    cvt(Wk, wkB, (size_t)NLAYER * CEMB * CEMB);
    cvt(Wv, wvB, (size_t)NLAYER * CEMB * CEMB);
    cvt(Wo, woB, (size_t)NLAYER * CEMB * CEMB);
    cvt(W1, w1B, (size_t)NLAYER * CEMB * FFDIM);
    cvt(W2, w2B, (size_t)NLAYER * FFDIM * CEMB);
    cvt(Wh, whB, (size_t)CEMB * VOCAB);

    auto gemm = [&](const unsigned short* A, const unsigned short* Bw, const float* bias,
                    float* outf, unsigned short* outb, int M, int N, int K, int addRes, int relu) {
        dim3 g((N + 63) / 64, M / 256);
        gemm_bf16_kernel<<<g, 256, 0, stream>>>(A, Bw, bias, outf, outb, M, N, K, addRes, relu);
    };

    embed_kernel<<<BT, 256, 0, stream>>>(idx, tok, pos, xf);

    for (int l = 0; l < NLAYER; ++l) {
        const size_t wcc = (size_t)l * CEMB * CEMB;
        // ln1 -> bf16
        ln_kernel<<<BT, 128, 0, stream>>>(xf, ln1g + (size_t)l * CEMB, ln1b + (size_t)l * CEMB, hb);
        // q, k, v projections (no bias)
        gemm(hb, wqB + wcc, nullptr, nullptr, qbf, BT, CEMB, CEMB, 0, 0);
        gemm(hb, wkB + wcc, nullptr, nullptr, kbf, BT, CEMB, CEMB, 0, 0);
        gemm(hb, wvB + wcc, nullptr, nullptr, vbf, BT, CEMB, CEMB, 0, 0);
        // attention
        attn_kernel<<<dim3(TSEQ / 32, NHEAD, BATCH), 256, 0, stream>>>(qbf, kbf, vbf, obf);
        // output projection + residual
        gemm(obf, woB + wcc, bo + (size_t)l * CEMB, xf, nullptr, BT, CEMB, CEMB, 1, 0);
        // ln2 -> bf16
        ln_kernel<<<BT, 128, 0, stream>>>(xf, ln2g + (size_t)l * CEMB, ln2b + (size_t)l * CEMB, hb);
        // MLP: relu(h @ W1 + b1) @ W2 + b2, residual
        gemm(hb, w1B + (size_t)l * CEMB * FFDIM, b1 + (size_t)l * FFDIM, nullptr, a1b, BT, FFDIM, CEMB, 0, 1);
        gemm(a1b, w2B + (size_t)l * FFDIM * CEMB, b2 + (size_t)l * CEMB, xf, nullptr, BT, CEMB, FFDIM, 1, 0);
    }

    // final LN + head -> logits (fp32, directly into d_out)
    ln_kernel<<<BT, 128, 0, stream>>>(xf, lnfg, lnfb, hb);
    float* logits = (float*)d_out;
    gemm(hb, whB, bh, logits, nullptr, BT, VOCAB, CEMB, 0, 0);

    // loss -> d_out[BT*VOCAB]
    loss_init_kernel<<<1, 32, 0, stream>>>(lossAcc);
    loss_row_kernel<<<BT / 256, 256, 0, stream>>>(logits, tgt, lossAcc, BT);
    loss_final_kernel<<<1, 32, 0, stream>>>(lossAcc, logits + (size_t)BT * VOCAB);
}
